// CustomDistanceTransformerLayer_63642825392388
// MI455X (gfx1250) — matrix operand order repaired and checked by reading, not compiled, not run on hardware
//
#include <hip/hip_runtime.h>

#define N_TOK 8192
#define EMB   512
#define DFF   2048
#define CB    128   // attention column block per iteration (8 waves x 16)

typedef _Float16 v8h  __attribute__((ext_vector_type(8)));
typedef _Float16 v16h __attribute__((ext_vector_type(16)));
typedef float    v8f  __attribute__((ext_vector_type(8)));
typedef unsigned int v4u __attribute__((ext_vector_type(4)));
typedef int      v4i  __attribute__((ext_vector_type(4)));
typedef int      v8i  __attribute__((ext_vector_type(8)));

// ---------------------------------------------------------------------------
// WMMA helpers (CDNA5 16x16x32 f16 -> f32), contiguous k = 16*(lane/16) + s fragment packing
// ---------------------------------------------------------------------------
__device__ __forceinline__ v8f wmma_f16(v16h a, v16h b, v8f c) {
    return __builtin_amdgcn_wmma_f32_16x16x32_f16(
        /*neg_a=*/false, a, /*neg_b=*/false, b,
        /*c_mod=*/(short)0, c, /*reuse_a=*/false, /*reuse_b=*/false);
}

// A fragment 16x32 (MxK), source row-major [m][k], leading dim `ld` (halves).
__device__ __forceinline__ v16h load_a_frag(const _Float16* base, int ld, int lane) {
    const _Float16* p = base + (size_t)(lane & 15) * ld + ((lane >> 4) << 4);
    v8h lo = *(const v8h*)(p);        // K k0..k0+7
    v8h hi = *(const v8h*)(p + 8);    // K k0+8..k0+15
    return __builtin_shufflevector(lo, hi, 0,1,2,3,4,5,6,7,8,9,10,11,12,13,14,15);
}

// B fragment 32x16 (KxN), source laid out [n][k], leading dim `ld` (halves).
__device__ __forceinline__ v16h load_b_frag(const _Float16* base, int ld, int lane) {
    const _Float16* p = base + (size_t)(lane & 15) * ld + ((lane >> 4) << 4);
    return *(const v16h*)(p);         // 32B contiguous
}

// ---------------------------------------------------------------------------
// Tensor Data Mover: 2-D tile load (f16 elements) into LDS.
// D# packing per CDNA5 ISA 8.3/8.4: group0 = {count/type/lds/global addr},
// group1 = {data_size, tensor dims, tile dims, dim0 stride}; groups 2/3 zero
// (tile_dim2 == 0 -> dims beyond 2 unused). Issue from ONE wave; EXEC ignored.
// This toolchain exposes the 6-arg builtin (extra int32x8 group, zero-filled).
// ---------------------------------------------------------------------------
__device__ __forceinline__ void tdm_load_2d(unsigned lds_byte_off,
                                            const _Float16* gptr,
                                            unsigned tile_w, unsigned tile_h,
                                            unsigned row_stride_elems,
                                            unsigned tensor_w, unsigned tensor_h) {
    unsigned long long ga = (unsigned long long)(size_t)gptr;
    v4u g0;
    g0.x = 1u;                                           // count=1, user desc
    g0.y = lds_byte_off;                                 // lds_addr
    g0.z = (unsigned)(ga & 0xFFFFFFFFull);               // global_addr[31:0]
    g0.w = (unsigned)((ga >> 32) & 0x01FFFFFFull)        // global_addr[56:32]
           | (2u << 30);                                 // type=2 ("image")
    v8i g1;
    g1[0] = 0x00010000;                                  // data_size=1 (2 bytes)
    g1[1] = (int)((tensor_w & 0xFFFFu) << 16);           // tensor_dim0[15:0]
    g1[2] = (int)(((tensor_w >> 16) & 0xFFFFu)           // tensor_dim0[31:16]
           | ((tensor_h & 0xFFFFu) << 16));              // tensor_dim1[15:0]
    g1[3] = (int)(((tensor_h >> 16) & 0xFFFFu)           // tensor_dim1[31:16]
           | ((tile_w & 0xFFFFu) << 16));                // tile_dim0
    g1[4] = (int)(tile_h & 0xFFFFu);                     // tile_dim1 (tile_dim2=0)
    g1[5] = (int)row_stride_elems;                       // tensor_dim0_stride[31:0]
    g1[6] = 0;                                           // stride hi, dim1_stride lo
    g1[7] = 0;
    v4i gz4 = {0, 0, 0, 0};
    v8i gz8 = {0, 0, 0, 0, 0, 0, 0, 0};
    __builtin_amdgcn_tensor_load_to_lds(g0, g1, gz4, gz4, gz8, /*cpol=*/0);
}

// ---------------------------------------------------------------------------
// LayerNorm: one block per row (256 threads, 2 elements each), f32 -> f16
// ---------------------------------------------------------------------------
__global__ __launch_bounds__(256) void ln_kernel(const float* __restrict__ x,
                                                 const float* __restrict__ g,
                                                 const float* __restrict__ b,
                                                 _Float16* __restrict__ out) {
    __shared__ float ssum[8];
    __shared__ float svar[8];
    const int row = blockIdx.x, tid = threadIdx.x;
    const float* xr = x + (size_t)row * EMB;
    float v0 = xr[tid], v1 = xr[tid + 256];

    float s = v0 + v1;
#pragma unroll
    for (int o = 16; o > 0; o >>= 1) s += __shfl_xor(s, o);
    if ((tid & 31) == 0) ssum[tid >> 5] = s;
    __syncthreads();
    float tot = 0.f;
#pragma unroll
    for (int w = 0; w < 8; ++w) tot += ssum[w];
    const float mean = tot * (1.f / (float)EMB);

    const float d0 = v0 - mean, d1 = v1 - mean;
    float vs = d0 * d0 + d1 * d1;
#pragma unroll
    for (int o = 16; o > 0; o >>= 1) vs += __shfl_xor(vs, o);
    if ((tid & 31) == 0) svar[tid >> 5] = vs;
    __syncthreads();
    float vtot = 0.f;
#pragma unroll
    for (int w = 0; w < 8; ++w) vtot += svar[w];
    const float inv = rsqrtf(vtot * (1.f / (float)EMB) + 1e-5f);

    out[(size_t)row * EMB + tid]       = (_Float16)(d0 * inv * g[tid] + b[tid]);
    out[(size_t)row * EMB + 256 + tid] = (_Float16)(d1 * inv * g[tid + 256] + b[tid + 256]);
}

// ---------------------------------------------------------------------------
// f32 -> f16 conversion (weights)
// ---------------------------------------------------------------------------
__global__ __launch_bounds__(256) void cvt_kernel(const float* __restrict__ in,
                                                  _Float16* __restrict__ out, int n) {
    int i = blockIdx.x * 256 + threadIdx.x;
    if (i < n) out[i] = (_Float16)in[i];
}

// ---------------------------------------------------------------------------
// Fused flash attention + additive bias + residual.
// Block: 16 query rows, 256 threads (8 waves). K/V tiles DMA'd by the TDM.
// x2 = x + softmax(nx.nx^T/sqrt(E) + spi) @ nx
// ---------------------------------------------------------------------------
__global__ __launch_bounds__(256) void attn_kernel(const float* __restrict__ x,
                                                   const float* __restrict__ spi,
                                                   const _Float16* __restrict__ nx,
                                                   float* __restrict__ x2) {
    extern __shared__ char smem[];
    _Float16* Qs  = (_Float16*)smem;                   // 16  x 512 halves (16KB)
    _Float16* KVs = Qs  + 16 * EMB;                    // 128 x 512 [row][k] (128KB)
    _Float16* KVt = KVs + CB * EMB;                    // 512 x 128 [n][k]   (128KB)
    float*    Sp  = (float*)(KVt + EMB * CB);          // 16 x 128 scores    (8KB)
    _Float16* Ph  = (_Float16*)(Sp + 16 * CB);         // 16 x 128 probs f16 (4KB)
    float*    rowm = (float*)(Ph + 16 * CB);           // running max [16]
    float*    rowl = rowm + 16;                        // running sum [16]
    float*    rowc = rowl + 16;                        // rescale     [16]
    float*    wmax = rowc + 16;                        // per-wave row max [8][16]

    const int tid  = threadIdx.x;
    const int wave = tid >> 5;
    const int lane = tid & 31;
    const int row0 = blockIdx.x * 16;
    const int hw   = lane >> 4;
    const int ncol = lane & 15;
    const float scale = 0.044194173824159216f;  // 1/sqrt(512)

    const unsigned qs_off  = (unsigned)(size_t)Qs;
    const unsigned kvs_off = (unsigned)(size_t)KVs;

    // ---- TDM: stage Q tile (16 x 512 f16) ----
    if (tid < 32) {
        tdm_load_2d(qs_off, nx + (size_t)row0 * EMB, EMB, 16, EMB, EMB, N_TOK);
        __builtin_amdgcn_s_wait_tensorcnt(0);
    }
    if (tid < 16) { rowm[tid] = -1e30f; rowl[tid] = 0.f; }
    __syncthreads();

    // ---- Q fragments: 16 K-chunks of 32 ----
    v16h qf[16];
#pragma unroll
    for (int c = 0; c < 16; ++c) qf[c] = load_a_frag(Qs + 32 * c, EMB, lane);

    const v8f vzero = {0.f,0.f,0.f,0.f,0.f,0.f,0.f,0.f};
    v8f acc[4] = {vzero, vzero, vzero, vzero};   // 16 rows x 64 output cols

    for (int jc = 0; jc < N_TOK; jc += CB) {
        // (A) TDM DMA of 128x512 K/V tile into KVs (wave 0 issues + waits)
        if (tid < 32) {
            tdm_load_2d(kvs_off, nx + (size_t)jc * EMB, EMB, CB, EMB, EMB, N_TOK);
            __builtin_amdgcn_s_wait_tensorcnt(0);
        }
        __syncthreads();
        // transposed copy for the P@V B-operand (TDM cannot transpose)
        for (int i = tid; i < CB * EMB / 8; i += 256) {
            int r = i >> 6, k8 = (i & 63) << 3;
            v8h v = *(const v8h*)&KVs[r * EMB + k8];
#pragma unroll
            for (int e = 0; e < 8; ++e) KVt[(size_t)(k8 + e) * CB + r] = v[e];
        }
        __syncthreads();

        // (B) scores: wave's 16x16 tile at column base wave*16, full K=512
        v8f s = vzero;
#pragma unroll
        for (int c = 0; c < 16; ++c) {
            v16h bf = load_b_frag(KVs + (size_t)(wave * 16) * EMB + 32 * c, EMB, lane);
            s = wmma_f16(qf[c], bf, s);
        }
        {
            const int gcol = jc + wave * 16 + ncol;
#pragma unroll
            for (int r = 0; r < 8; ++r) {
                int m = r + (hw << 3);
                float sv = s[r] * scale + spi[(size_t)(row0 + m) * N_TOK + gcol];
                Sp[m * CB + wave * 16 + ncol] = sv;
                float mx = sv;
                mx = fmaxf(mx, __shfl_xor(mx, 1));
                mx = fmaxf(mx, __shfl_xor(mx, 2));
                mx = fmaxf(mx, __shfl_xor(mx, 4));
                mx = fmaxf(mx, __shfl_xor(mx, 8));
                if (ncol == 0) wmax[wave * 16 + m] = mx;
            }
        }
        __syncthreads();

        // (C) combine per-wave maxima, update running softmax state
        if (tid < 16) {
            float bm = wmax[tid];
#pragma unroll
            for (int w = 1; w < 8; ++w) bm = fmaxf(bm, wmax[w * 16 + tid]);
            float mo = rowm[tid];
            float mn = fmaxf(mo, bm);
            float corr = __expf(mo - mn);
            rowm[tid] = mn;
            rowc[tid] = corr;
            rowl[tid] *= corr;
        }
        __syncthreads();

        // (D) exponentiate block, f16 copy for WMMA
        for (int i = tid; i < 16 * CB; i += 256) {
            int m = i >> 7;
            float p = __expf(Sp[i] - rowm[m]);
            Sp[i] = p;
            Ph[i] = (_Float16)p;
        }
        __syncthreads();

        // (E) row sums, accumulator rescale, P @ V
        if (tid < 16) {
            float sum = 0.f;
            for (int j = 0; j < CB; ++j) sum += Sp[tid * CB + j];
            rowl[tid] += sum;
        }
#pragma unroll
        for (int t = 0; t < 4; ++t) {
#pragma unroll
            for (int r = 0; r < 8; ++r) acc[t][r] *= rowc[r + (hw << 3)];
        }
#pragma unroll
        for (int c2 = 0; c2 < 4; ++c2) {
            v16h af = load_a_frag(Ph + 32 * c2, CB, lane);
#pragma unroll
            for (int t = 0; t < 4; ++t) {
                int nb = wave * 64 + t * 16;
                v16h bf = load_b_frag(KVt + (size_t)nb * CB + 32 * c2, CB, lane);
                acc[t] = wmma_f16(af, bf, acc[t]);
            }
        }
        __syncthreads();
    }

    // ---- epilogue: normalize + residual ----
#pragma unroll
    for (int t = 0; t < 4; ++t) {
        int n = wave * 64 + t * 16 + ncol;
#pragma unroll
        for (int r = 0; r < 8; ++r) {
            int m = r + (hw << 3);
            float o = acc[t][r] / rowl[m];
            size_t idx = (size_t)(row0 + m) * EMB + n;
            x2[idx] = x[idx] + o;
        }
    }
}

// ---------------------------------------------------------------------------
// FFN GEMM 1: act = relu(h @ W1 + bb1)   [8192,512]x[512,2048] -> f16
// Block tile 64x128, A tile DMA'd by TDM, B tile staged transposed.
// ---------------------------------------------------------------------------
__global__ __launch_bounds__(256) void ffn1_kernel(const _Float16* __restrict__ H,
                                                   const _Float16* __restrict__ W1h,
                                                   const float* __restrict__ bb1,
                                                   _Float16* __restrict__ act) {
    __shared__ _Float16 As[64 * 32];     // [m][k]
    __shared__ _Float16 Bs[128 * 32];    // [n][k]
    const int tid = threadIdx.x, lane = tid & 31, wave = tid >> 5;
    const int wm = wave >> 2, wn = wave & 3;
    const int gm0 = blockIdx.y * 64, gn0 = blockIdx.x * 128;
    const int hw = lane >> 4, ncol = lane & 15;
    const unsigned as_off = (unsigned)(size_t)As;

    const v8f vzero = {0.f,0.f,0.f,0.f,0.f,0.f,0.f,0.f};
    v8f acc[2][2] = {{vzero, vzero}, {vzero, vzero}};

    for (int kc = 0; kc < EMB; kc += 32) {
        if (tid < 32) {   // TDM: A tile 64 rows x 32 halves, row stride EMB
            tdm_load_2d(as_off, &H[(size_t)gm0 * EMB + kc], 32, 64, EMB, EMB, N_TOK);
            __builtin_amdgcn_s_wait_tensorcnt(0);
        }
        {   // stage B transposed: each thread moves one (k, 16n) strip
            int k = tid >> 3, n0 = (tid & 7) << 4;
            const _Float16* src = &W1h[(size_t)(kc + k) * DFF + gn0 + n0];
#pragma unroll
            for (int e = 0; e < 16; ++e) Bs[(n0 + e) * 32 + k] = src[e];
        }
        __syncthreads();
        v16h a0 = load_a_frag(As + (wm * 32) * 32, 32, lane);
        v16h a1 = load_a_frag(As + (wm * 32 + 16) * 32, 32, lane);
        v16h b0 = load_b_frag(Bs + (wn * 32) * 32, 32, lane);
        v16h b1 = load_b_frag(Bs + (wn * 32 + 16) * 32, 32, lane);
        acc[0][0] = wmma_f16(a0, b0, acc[0][0]);
        acc[0][1] = wmma_f16(a0, b1, acc[0][1]);
        acc[1][0] = wmma_f16(a1, b0, acc[1][0]);
        acc[1][1] = wmma_f16(a1, b1, acc[1][1]);
        __syncthreads();
    }

#pragma unroll
    for (int mt = 0; mt < 2; ++mt) {
#pragma unroll
        for (int nt = 0; nt < 2; ++nt) {
            int n = gn0 + wn * 32 + nt * 16 + ncol;
            float bias = bb1[n];
#pragma unroll
            for (int r = 0; r < 8; ++r) {
                int m = gm0 + wm * 32 + mt * 16 + r + (hw << 3);
                float v = acc[mt][nt][r] + bias;
                act[(size_t)m * DFF + n] = (_Float16)fmaxf(v, 0.f);
            }
        }
    }
}

// ---------------------------------------------------------------------------
// FFN GEMM 2 + residual: out = x2 + act @ W2 + bb2  [8192,2048]x[2048,512]
// ---------------------------------------------------------------------------
__global__ __launch_bounds__(256) void ffn2_kernel(const _Float16* __restrict__ A,
                                                   const _Float16* __restrict__ W2h,
                                                   const float* __restrict__ bb2,
                                                   const float* __restrict__ x2,
                                                   float* __restrict__ out) {
    __shared__ _Float16 As[64 * 32];
    __shared__ _Float16 Bs[128 * 32];
    const int tid = threadIdx.x, lane = tid & 31, wave = tid >> 5;
    const int wm = wave >> 2, wn = wave & 3;
    const int gm0 = blockIdx.y * 64, gn0 = blockIdx.x * 128;
    const int hw = lane >> 4, ncol = lane & 15;
    const unsigned as_off = (unsigned)(size_t)As;

    const v8f vzero = {0.f,0.f,0.f,0.f,0.f,0.f,0.f,0.f};
    v8f acc[2][2] = {{vzero, vzero}, {vzero, vzero}};

    for (int kc = 0; kc < DFF; kc += 32) {
        if (tid < 32) {
            tdm_load_2d(as_off, &A[(size_t)gm0 * DFF + kc], 32, 64, DFF, DFF, N_TOK);
            __builtin_amdgcn_s_wait_tensorcnt(0);
        }
        {
            int k = tid >> 3, n0 = (tid & 7) << 4;
            const _Float16* src = &W2h[(size_t)(kc + k) * EMB + gn0 + n0];
#pragma unroll
            for (int e = 0; e < 16; ++e) Bs[(n0 + e) * 32 + k] = src[e];
        }
        __syncthreads();
        v16h a0 = load_a_frag(As + (wm * 32) * 32, 32, lane);
        v16h a1 = load_a_frag(As + (wm * 32 + 16) * 32, 32, lane);
        v16h b0 = load_b_frag(Bs + (wn * 32) * 32, 32, lane);
        v16h b1 = load_b_frag(Bs + (wn * 32 + 16) * 32, 32, lane);
        acc[0][0] = wmma_f16(a0, b0, acc[0][0]);
        acc[0][1] = wmma_f16(a0, b1, acc[0][1]);
        acc[1][0] = wmma_f16(a1, b0, acc[1][0]);
        acc[1][1] = wmma_f16(a1, b1, acc[1][1]);
        __syncthreads();
    }

#pragma unroll
    for (int mt = 0; mt < 2; ++mt) {
#pragma unroll
        for (int nt = 0; nt < 2; ++nt) {
            int n = gn0 + wn * 32 + nt * 16 + ncol;
            float bias = bb2[n];
#pragma unroll
            for (int r = 0; r < 8; ++r) {
                int m = gm0 + wm * 32 + mt * 16 + r + (hw << 3);
                size_t idx = (size_t)m * EMB + n;
                out[idx] = x2[idx] + acc[mt][nt][r] + bias;
            }
        }
    }
}

// ---------------------------------------------------------------------------
// Host launcher
// ---------------------------------------------------------------------------
extern "C" void kernel_launch(void* const* d_in, const int* in_sizes, int n_in,
                              void* d_out, int out_size, void* d_ws, size_t ws_size,
                              hipStream_t stream) {
    (void)in_sizes; (void)n_in; (void)out_size; (void)ws_size;
    const float* x   = (const float*)d_in[0];
    // d_in[1] = edge_weights: accepted but unused by the reference forward.
    const float* spi = (const float*)d_in[2];
    const float* g1  = (const float*)d_in[3];
    const float* b1  = (const float*)d_in[4];
    const float* g2  = (const float*)d_in[5];
    const float* b2  = (const float*)d_in[6];
    const float* W1  = (const float*)d_in[7];
    const float* bb1 = (const float*)d_in[8];
    const float* W2  = (const float*)d_in[9];
    const float* bb2 = (const float*)d_in[10];
    float* out = (float*)d_out;

    char* ws = (char*)d_ws;
    _Float16* nx_h  = (_Float16*)(ws);                            //  8 MB
    float*    x2    = (float*)   (ws + (size_t)8  * 1024 * 1024); // 16 MB
    _Float16* h_h   = (_Float16*)(ws + (size_t)24 * 1024 * 1024); //  8 MB
    _Float16* act_h = (_Float16*)(ws + (size_t)32 * 1024 * 1024); // 32 MB
    _Float16* W1h   = (_Float16*)(ws + (size_t)64 * 1024 * 1024); //  2 MB
    _Float16* W2h   = (_Float16*)(ws + (size_t)66 * 1024 * 1024); //  2 MB

    ln_kernel<<<dim3(N_TOK), dim3(256), 0, stream>>>(x, g1, b1, nx_h);

    cvt_kernel<<<dim3((EMB * DFF) / 256), dim3(256), 0, stream>>>(W1, W1h, EMB * DFF);
    cvt_kernel<<<dim3((DFF * EMB) / 256), dim3(256), 0, stream>>>(W2, W2h, DFF * EMB);

    const size_t attn_smem =
        (size_t)16 * EMB * 2 + (size_t)CB * EMB * 2 + (size_t)EMB * CB * 2 +
        (size_t)16 * CB * 4 + (size_t)16 * CB * 2 + (size_t)(16 + 16 + 16 + 128) * 4;
    attn_kernel<<<dim3(N_TOK / 16), dim3(256), attn_smem, stream>>>(x, spi, nx_h, x2);

    ln_kernel<<<dim3(N_TOK), dim3(256), 0, stream>>>(x2, g2, b2, h_h);

    ffn1_kernel<<<dim3(DFF / 128, N_TOK / 64), dim3(256), 0, stream>>>(h_h, W1h, bb1, act_h);

    ffn2_kernel<<<dim3(EMB / 128, N_TOK / 64), dim3(256), 0, stream>>>(act_h, W2h, bb2, x2, out);
}
